// HyenaModule_33062658244720
// MI455X (gfx1250) — matrix operand order repaired and checked by reading, not compiled, not run on hardware
//
#include <hip/hip_runtime.h>

// ---------------------------------------------------------------------------
// Types / helpers
// ---------------------------------------------------------------------------
typedef __attribute__((ext_vector_type(16))) __bf16 v16bf;
typedef __attribute__((ext_vector_type(2)))  __bf16 v2bf;
typedef __attribute__((ext_vector_type(8)))  float  v8f;
typedef unsigned short bf16_t;
typedef unsigned int   u32;

union Frag {
    v16bf  v;
    u32    u[8];
    bf16_t h[16];
};

// Native bf16 converts: CDNA5 is bf16-native.
__device__ __forceinline__ bf16_t f2bf(float f) {
    union { __bf16 h; bf16_t u; } a; a.h = (__bf16)f;
    return a.u;
}
__device__ __forceinline__ float bf2f(bf16_t h) {
    union { bf16_t u; __bf16 h; } a; a.u = h;
    return (float)a.h;
}
// Convert a float pair into one packed dword: single v_cvt_pk_bf16_f32.
#if __has_builtin(__builtin_amdgcn_cvt_pk_bf16_f32)
__device__ __forceinline__ u32 pack_bf16x2(float lo, float hi) {
    union { v2bf v; u32 u; } p;
    p.v = __builtin_amdgcn_cvt_pk_bf16_f32(lo, hi);
    return p.u;
}
#elif defined(__AMDGCN__)
__device__ __forceinline__ u32 pack_bf16x2(float lo, float hi) {
    u32 r;
    asm("v_cvt_pk_bf16_f32 %0, %1, %2" : "=v"(r) : "v"(lo), "v"(hi));
    return r;
}
#else
__device__ __forceinline__ u32 pack_bf16x2(float lo, float hi) {
    union { __bf16 h[2]; u32 u; } p;
    p.h[0] = (__bf16)lo; p.h[1] = (__bf16)hi;
    return p.u;
}
#endif
__device__ __forceinline__ u32 pack_bf16x2(bf16_t lo, bf16_t hi) {
    return (u32)lo | ((u32)hi << 16);
}
__device__ __forceinline__ bf16_t to_bf16s(float f)  { return f2bf(f); }
__device__ __forceinline__ bf16_t to_bf16s(bf16_t h) { return h; }

__device__ __forceinline__ void store_out(float*  p, size_t i, float v) { p[i] = v; }
__device__ __forceinline__ void store_out(bf16_t* p, size_t i, float v) { p[i] = f2bf(v); }

// Problem constants
#define CB    16
#define CL    1024
#define CDIN  512
#define CD    1024
#define C3D   3072
#define CM    (CB * CL)        // 16384 rows

// ---------------------------------------------------------------------------
// bc = b_in + bp @ W_in
// ---------------------------------------------------------------------------
__global__ __launch_bounds__(256) void bias_fuse_kernel(
    const float* __restrict__ bp, const float* __restrict__ W_in,
    const float* __restrict__ b_in, float* __restrict__ bc)
{
    int j = blockIdx.x * 256 + threadIdx.x;
    if (j >= C3D) return;
    float s = b_in[j];
    for (int d = 0; d < CD; ++d) s += bp[d] * W_in[(size_t)d * C3D + j];
    bc[j] = s;
}

// ---------------------------------------------------------------------------
// Generic bf16-WMMA GEMM: C[M,N] = A[M,K] @ B[K,N] (+ bias[N])
// Double-buffered LDS; block tile 128x128x32, 256 threads = 8 waves (4x2).
// M,N multiples of 128; K multiple of 32 (true for all uses here).
// B staging: each thread owns 2 adjacent k-rows x 8 n-cols so the (k,k+1)
// pair packs into one dword (v_cvt_pk_bf16_f32 + ds_store_b32).
// ---------------------------------------------------------------------------
#define BM 128
#define BN 128
#define BK 32
#define LDPA 48   // padded LDS row stride so frag reads are 16B-aligned b128s

template <typename AT, typename BT, typename OT, bool BIAS>
__global__ __launch_bounds__(256) void gemm_wmma_kernel(
    const AT* __restrict__ A, const BT* __restrict__ Bm,
    const float* __restrict__ bias, OT* __restrict__ C,
    int M, int N, int K)
{
    __shared__ bf16_t sA [2][BM][LDPA];   // row-major A tiles
    __shared__ bf16_t sBt[2][BN][LDPA];   // transposed B tiles: [n][k]

    const int tid  = threadIdx.x;
    const int lane = tid & 31;
    const int wave = tid >> 5;
    const int wm   = wave & 3;        // 0..3 along M
    const int wn   = wave >> 2;       // 0..1 along N
    const int hf   = lane >> 4;       // wave half
    const int mn   = lane & 15;       // row (A) / col (B,C) within tile

    const int m0 = blockIdx.y * BM;
    const int n0 = blockIdx.x * BN;

    v8f acc[2][4];
#pragma unroll
    for (int tm = 0; tm < 2; ++tm)
#pragma unroll
        for (int tn = 0; tn < 4; ++tn)
#pragma unroll
            for (int e = 0; e < 8; ++e) acc[tm][tn][e] = 0.0f;

    const int ar  = tid >> 1,       ac0 = (tid & 1) * 16;  // A: (row, 16 cols)
    const int br2 = (tid >> 4) * 2, bn8 = (tid & 15) * 8;  // B: (2 k-rows, 8 cols)

    // stage tile 0 into buffer 0
    {
        const size_t ab  = (size_t)(m0 + ar) * K + ac0;
        const size_t bb0 = (size_t)(br2    ) * N + n0 + bn8;
        const size_t bb1 = (size_t)(br2 + 1) * N + n0 + bn8;
#pragma unroll
        for (int i = 0; i < 16; i += 2)
            *(u32*)&sA[0][ar][ac0 + i] = pack_bf16x2(A[ab + i], A[ab + i + 1]);
#pragma unroll
        for (int i = 0; i < 8; ++i)
            *(u32*)&sBt[0][bn8 + i][br2] = pack_bf16x2(Bm[bb0 + i], Bm[bb1 + i]);
    }
    __syncthreads();

    const int nk = K / BK;
    for (int kt = 0; kt < nk; ++kt) {
        const int cur = kt & 1, nxt = cur ^ 1;

        // prefetch next K-tile from global into registers (overlaps with WMMA)
        AT ra[16];
        BT rb0[8], rb1[8];
        const bool more = (kt + 1 < nk);
        if (more) {
            const int kb = (kt + 1) * BK;
            const size_t ab  = (size_t)(m0 + ar) * K + kb + ac0;
            const size_t bb0 = (size_t)(kb + br2    ) * N + n0 + bn8;
            const size_t bb1 = (size_t)(kb + br2 + 1) * N + n0 + bn8;
#pragma unroll
            for (int i = 0; i < 16; ++i) ra[i] = A[ab + i];
#pragma unroll
            for (int i = 0; i < 8; ++i)  rb0[i] = Bm[bb0 + i];
#pragma unroll
            for (int i = 0; i < 8; ++i)  rb1[i] = Bm[bb1 + i];
        }

        // compute on current buffer
        Frag a[2], b[4];
#pragma unroll
        for (int tm = 0; tm < 2; ++tm) {
            const int row = wm * 32 + tm * 16 + mn;
#pragma unroll
            for (int j = 0; j < 8; ++j) {
                const int off = hf * 16 + 2 * j;
                a[tm].u[j] = *(const u32*)&sA[cur][row][off];
            }
        }
#pragma unroll
        for (int tn = 0; tn < 4; ++tn) {
            const int col = wn * 64 + tn * 16 + mn;
            const int kk  = hf * 16;
#pragma unroll
            for (int j = 0; j < 8; ++j)
                b[tn].u[j] = *(const u32*)&sBt[cur][col][kk + 2 * j];
        }
#pragma unroll
        for (int tm = 0; tm < 2; ++tm)
#pragma unroll
            for (int tn = 0; tn < 4; ++tn)
                acc[tm][tn] = __builtin_amdgcn_wmma_f32_16x16x32_bf16(
                    false, a[tm].v, false, b[tn].v, (short)0, acc[tm][tn], false, false);

        // commit prefetched tile to the other buffer (packed converts)
        if (more) {
#pragma unroll
            for (int i = 0; i < 16; i += 2)
                *(u32*)&sA[nxt][ar][ac0 + i] = pack_bf16x2(ra[i], ra[i + 1]);
#pragma unroll
            for (int i = 0; i < 8; ++i)
                *(u32*)&sBt[nxt][bn8 + i][br2] = pack_bf16x2(rb0[i], rb1[i]);
        }
        __syncthreads();
    }

#pragma unroll
    for (int tm = 0; tm < 2; ++tm)
#pragma unroll
        for (int tn = 0; tn < 4; ++tn) {
            const int col = n0 + wn * 64 + tn * 16 + mn;
            float badd = BIAS ? bias[col] : 0.0f;
#pragma unroll
            for (int r = 0; r < 8; ++r) {
                const int row = m0 + wm * 32 + tm * 16 + r + 8 * hf;
                store_out(C, (size_t)row * N + col, acc[tm][tn][r] + badd);
            }
        }
}

// ---------------------------------------------------------------------------
// Depthwise causal conv-3 + gate + transpose (branch-free boundaries):
//   g1t[d][l][b] = convV * convX2 ; x1t[d][l][b] = convX1   (bf16)
// ---------------------------------------------------------------------------
__global__ __launch_bounds__(256) void shortconv_gate_kernel(
    const bf16_t* __restrict__ u3, const float* __restrict__ sw,
    const float* __restrict__ sb, bf16_t* __restrict__ g1t,
    bf16_t* __restrict__ x1t)
{
    const int d = blockIdx.x;
    const int b = threadIdx.x & 15;
    const int l = blockIdx.y * 16 + (threadIdx.x >> 4);

    float cv[3];
#pragma unroll
    for (int g = 0; g < 3; ++g) {
        const int c = g * CD + d;
        float acc = sb[c];
#pragma unroll
        for (int j = 0; j < 3; ++j) {
            const int li  = l - 2 + j;
            const int lic = li < 0 ? 0 : li;              // clamp: load always valid
            const float w = (li >= 0) ? sw[c * 3 + j] : 0.0f;  // select, not branch
            acc += w * bf2f(u3[((size_t)b * CL + lic) * C3D + c]);
        }
        cv[g] = acc;
    }
    const size_t o = (size_t)d * CL * CB + (size_t)l * CB + b;
    g1t[o] = f2bf(cv[2] * cv[1]);   // v * x2
    x1t[o] = f2bf(cv[0]);           // x1
}

// ---------------------------------------------------------------------------
// Causal long conv: lower-triangular Toeplitz x [L,16] per channel on WMMA.
//   y[t,b] = sum_{s<=t} k[t-s]*g[s,b] + Dbias*g[t,b]
// A-fragment dwords come pre-packed & REVERSED from sPrev so that the 8
// fragment dwords are read at ASCENDING LDS offsets -> ds_load_2addr pairs
// land directly in fragment register order.
// MODE 0: out[d][t][b] = y * gate[d][t][b] ;  MODE 1: out[b][t][d] = y
// ---------------------------------------------------------------------------
#define SBL   1032          // padded [b][s] stride (bf16 elems)
#define SPN   1056          // 1024 + 32 zero tail (reversed)

template <int MODE>
__global__ __launch_bounds__(256) void conv_toeplitz_kernel(
    const bf16_t* __restrict__ gt, const float* __restrict__ kf,
    const float* __restrict__ dbias, const bf16_t* __restrict__ gate,
    bf16_t* __restrict__ out)
{
    __shared__ bf16_t sB2[CB][SBL];   // [b][s] panel (33 KB)
    __shared__ u32    sPrev[SPN];     // reversed pre-packed filter (4.2 KB)

    const int d    = blockIdx.x;
    const int tid  = threadIdx.x;
    const int lane = tid & 31;
    const int wave = tid >> 5;
    const int hf   = lane >> 4;
    const int mn   = lane & 15;

    // --- stage the channel panel into LDS (b128 global loads) ---
    const bf16_t* gch = gt + (size_t)d * CL * CB;
    for (int i = tid; i < (CL * CB) / 8; i += 256) {
        uint4 vv = ((const uint4*)gch)[i];
        const int l  = i >> 1;
        const int b0 = (i & 1) * 8;
        u32 w[4] = {vv.x, vv.y, vv.z, vv.w};
#pragma unroll
        for (int j = 0; j < 4; ++j) {
            sB2[b0 + 2 * j    ][l] = (bf16_t)(w[j] & 0xFFFFu);
            sB2[b0 + 2 * j + 1][l] = (bf16_t)(w[j] >> 16);
        }
    }
    // --- reversed pre-packed filter table (branch-free clamped loads) ---
    for (int i = tid; i < SPN; i += 256) {
        const int vlo = 1023 - i;                 // low-half filter index
        const int vhi = 1022 - i;                 // high-half filter index
        const int clo = vlo < 0 ? 0 : vlo;
        const int chi = vhi < 0 ? 0 : vhi;
        const u32 lo  = (vlo >= 0) ? (u32)f2bf(kf[(size_t)clo * CD + d]) : 0u;
        const u32 hi  = (vhi >= 0) ? (u32)f2bf(kf[(size_t)chi * CD + d]) : 0u;
        sPrev[i] = lo | (hi << 16);
    }
    const float db = dbias[d];
    __syncthreads();

    // 8 waves; wave w handles t-tiles w, w+8, ..., w+56
    for (int ti = wave; ti < CL / 16; ti += 8) {
        const int T0 = ti * 16;
        v8f c;
#pragma unroll
        for (int e = 0; e < 8; ++e) c[e] = 0.0f;

        const int t   = T0 + mn;
        const int nch = (T0 + 16 + 31) >> 5;      // causal: only s <= t chunks
        for (int sc = 0; sc < nch; ++sc) {
            Frag a, b;
            // ascending-offset reads: base + {0,2,4,6} and base + {16,18,20,22}
            const int base = 1023 - t + sc * 32 + hf * 8;   // in [0, 1054]
#pragma unroll
            for (int j = 0; j < 8; ++j) {
                const int soff = ((j < 4) ? 0 : 16) + (j & 3) * 2;
                a.u[j] = sPrev[base + soff];               // ds_load_2addr pairs
            }
            const int kb = sc * 32 + hf * 8;
#pragma unroll
            for (int j = 0; j < 8; ++j)
                b.u[j] = *(const u32*)&sB2[mn][kb + ((j < 4) ? 0 : 16) + (j & 3) * 2]; // 2x ds_load_b128

            c = __builtin_amdgcn_wmma_f32_16x16x32_bf16(
                false, a.v, false, b.v, (short)0, c, false, false);
        }

        // epilogue: skip term + gate/store (straight-line; MODE is compile-time)
#pragma unroll
        for (int r = 0; r < 8; ++r) {
            const int tt = T0 + r + 8 * hf;
            float y = c[r] + db * bf2f(sB2[mn][tt]);
            if (MODE == 0) {
                const size_t o = (size_t)d * CL * CB + (size_t)tt * CB + mn;
                out[o] = f2bf(y * bf2f(gate[o]));
            } else {
                out[(size_t)mn * CL * CD + (size_t)tt * CD + d] = f2bf(y);
            }
        }
    }
}

// ---------------------------------------------------------------------------
// Host launcher
// ---------------------------------------------------------------------------
extern "C" void kernel_launch(void* const* d_in, const int* in_sizes, int n_in,
                              void* d_out, int out_size, void* d_ws, size_t ws_size,
                              hipStream_t stream)
{
    (void)in_sizes; (void)n_in; (void)out_size; (void)ws_size;

    const float* x       = (const float*)d_in[0];
    const float* Wp      = (const float*)d_in[1];
    const float* bp      = (const float*)d_in[2];
    const float* W_in    = (const float*)d_in[3];
    const float* b_in    = (const float*)d_in[4];
    const float* short_w = (const float*)d_in[5];
    const float* short_b = (const float*)d_in[6];
    const float* k_filt  = (const float*)d_in[7];
    const float* D_bias  = (const float*)d_in[8];
    const float* W_out   = (const float*)d_in[9];
    const float* b_out   = (const float*)d_in[10];
    float* out = (float*)d_out;

    char* ws = (char*)d_ws;
    bf16_t* Wc  = (bf16_t*)(ws + 0);            // 512*3072*2        =   3,145,728
    float*  bc  = (float*) (ws + 3145728);      // 3072*4            =      12,288
    bf16_t* u3  = (bf16_t*)(ws + 3158016);      // 16384*3072*2      = 100,663,296
    bf16_t* g1t = (bf16_t*)(ws + 103821312);    // 1024*1024*16*2    =  33,554,432
    bf16_t* x1t = (bf16_t*)(ws + 137375744);
    bf16_t* g2t = (bf16_t*)(ws + 170930176);
    bf16_t* y2  = (bf16_t*)(ws + 204484608);    // end = 238,039,040 bytes

    // 1) fused bias  bc = b_in + bp @ W_in
    bias_fuse_kernel<<<dim3((C3D + 255) / 256), 256, 0, stream>>>(bp, W_in, b_in, bc);

    // 2) fused weight  Wc = Wp @ W_in   (512 x 3072, K=1024) -> bf16
    gemm_wmma_kernel<float, float, bf16_t, false>
        <<<dim3(C3D / BN, CDIN / BM), 256, 0, stream>>>(Wp, W_in, nullptr, Wc,
                                                        CDIN, C3D, CD);

    // 3) fused projection  u3 = x @ Wc + bc   (16384 x 3072, K=512) -> bf16
    gemm_wmma_kernel<float, bf16_t, bf16_t, true>
        <<<dim3(C3D / BN, CM / BM), 256, 0, stream>>>(x, Wc, bc, u3,
                                                      CM, C3D, CDIN);

    // 4) depthwise causal conv-3 + gate + transpose to [D][L][B]
    shortconv_gate_kernel<<<dim3(CD, CL / 16), 256, 0, stream>>>(
        u3, short_w, short_b, g1t, x1t);

    // 5) stage-1 long conv (Toeplitz WMMA) + skip, gated by x1 -> g2t [D][L][B]
    conv_toeplitz_kernel<0><<<dim3(CD), 256, 0, stream>>>(
        g1t, k_filt, D_bias, x1t, g2t);

    // 6) stage-2 long conv + skip -> y2 [B][L][D]
    conv_toeplitz_kernel<1><<<dim3(CD), 256, 0, stream>>>(
        g2t, k_filt + (size_t)CL * CD, D_bias + CD, nullptr, y2);

    // 7) output projection  out = y2 @ W_out + b_out  (16384 x 1024, K=1024) fp32
    gemm_wmma_kernel<bf16_t, float, float, true>
        <<<dim3(CD / BN, CM / BM), 256, 0, stream>>>(y2, W_out, b_out, out,
                                                     CM, CD, CD);
}
